// VQVAEConv1D_60309930770501
// MI455X (gfx1250) — matrix operand order repaired and checked by reading, not compiled, not run on hardware
//
#include <hip/hip_runtime.h>
#include <stdint.h>
#include <stddef.h>

// ---------------------------------------------------------------------------
// CDNA5 (gfx1250) wave32 WMMA types
// ---------------------------------------------------------------------------
typedef __attribute__((ext_vector_type(16))) __bf16 v16bf;
typedef __attribute__((ext_vector_type(8)))  float  v8f;
typedef __attribute__((ext_vector_type(4)))  int    v4i;

union Frag16 { v16bf v; uint32_t u[8]; };

__device__ __forceinline__ unsigned short f2bf(float f) {
  union { float f; uint32_t u; } c; c.f = f;
  return (unsigned short)((c.u + 0x7FFFu + ((c.u >> 16) & 1u)) >> 16); // RNE
}

__device__ __forceinline__ v8f wmma_bf16(v16bf a, v16bf b, v8f c) {
  // D = A(16x32 bf16) * B(32x16 bf16) + C(16x16 f32)
  return __builtin_amdgcn_wmma_f32_16x16x32_bf16(false, a, false, b,
                                                 (short)0, c, false, false);
}

// ---------------------------------------------------------------------------
// Async global->LDS (gfx1250 GLOBAL_LOAD_ASYNC_TO_LDS_B128, ASYNCcnt) when
// the toolchain declares the builtins; otherwise vectorized uint4 fallback.
// Probe-learned signature: param0 = v4i* in AS1 (global), param1 = LDS side,
// then imm offset + imm cpol (matches llvm.amdgcn.global.load.async.to.lds).
// ---------------------------------------------------------------------------
#if __has_builtin(__builtin_amdgcn_global_load_async_to_lds_b128) && \
    __has_builtin(__builtin_amdgcn_s_wait_asynccnt)
#define USE_ASYNC_LDS 1
#else
#define USE_ASYNC_LDS 0
#endif

__device__ __forceinline__ void cp16_g2l(const unsigned short* g, unsigned short* l) {
#if USE_ASYNC_LDS
  __builtin_amdgcn_global_load_async_to_lds_b128(
      (__attribute__((address_space(1))) v4i*)g,
      (__attribute__((address_space(3))) v4i*)l, 0, 0);
#else
  *(uint4*)l = *(const uint4*)g;
#endif
}
__device__ __forceinline__ void cp_wait() {
#if USE_ASYNC_LDS
  __builtin_amdgcn_s_wait_asynccnt(0);
#endif
}

// ---------------------------------------------------------------------------
// Weight packer: build B matrix in [N][K] layout (Npad x Kpad, bf16) so tile
// staging is contiguous-K b128 loads.
//   GEMM K index kk = rr*CIN + cc  (rr = window-row, cc = input channel)
//   taps byte rr selects which kernel tap feeds window-row rr.
//   transposed=0 : w is (O, CIN, KT)  (Conv1d weight)
//   transposed=1 : w is (CIN, O, KT)  (ConvTranspose1d weight)
// ---------------------------------------------------------------------------
__global__ __launch_bounds__(256) void pack_weights(
    const float* __restrict__ w, unsigned short* __restrict__ Bp,
    int O, int Npad, int CIN, int KT, int KROWS, int Kpad, int taps, int transposed)
{
  int idx = blockIdx.x * 256 + threadIdx.x;
  if (idx >= Kpad * Npad) return;
  int n = idx / Kpad, kk = idx % Kpad;
  int rr = kk / CIN,  cc = kk % CIN;
  float v = 0.f;
  if (rr < KROWS && n < O) {
    int tap = (taps >> (8 * rr)) & 0xff;
    v = transposed ? w[((size_t)cc * O + n) * KT + tap]
                   : w[((size_t)n * CIN + cc) * KT + tap];
  }
  Bp[idx] = f2bf(v);
}

// ---------------------------------------------------------------------------
// Generic implicit-GEMM conv/deconv kernel.
//   Activations in (B, Tin, CIN) row-major.  Output row m = b*TLOC + t,
//   out[(b, OUT_T_MUL*t + OUT_T_OFF, n)] with Nreal columns.
//   A row m, K index k: window row rr=k/CIN, ch cc=k%CIN,
//   input time gt = TMUL*t + ROW_OFF + rr  (zero outside [0,TIN)).
// Block: 256 thr = 8 waves; C block tile 128(M) x 64(N); wave tile 32x32.
// LDS row stride 40 halfwords: 16B-aligned rows for b128 stores AND
// conflict-free fragment reads (20 dwords -> n*20 mod 64 distinct, 16 rows).
// ---------------------------------------------------------------------------
template<bool IN_F32, int CIN, int KROWS, int TMUL, int ROW_OFF,
         int OUT_T_MUL, int OUT_T_OFF, int TIN, int TLOC>
__global__ __launch_bounds__(256) void conv_gemm(
    const void* __restrict__ in_, const unsigned short* __restrict__ Bp,
    float* __restrict__ out, const float* __restrict__ bias,
    int Nreal, int Kpad)
{
  static_assert(IN_F32 || (CIN % 32) == 0, "bf16 fast path needs 32 | CIN");
  __shared__ unsigned short As[128 * 40];
  __shared__ unsigned short Bs[64 * 40];   // [n][k]

  const int tid  = threadIdx.x;
  const int lane = tid & 31;
  const int wave = tid >> 5;
  const int wm   = wave >> 1;     // 0..3
  const int wn   = wave & 1;      // 0..1
  const int half = lane >> 4;
  const int l16  = lane & 15;

  const int m0 = blockIdx.x * 128;
  const int n0 = blockIdx.y * 64;
  constexpr int K_REAL = KROWS * CIN;

  const float*          inF = (const float*)in_;
  const unsigned short* inH = (const unsigned short*)in_;

  v8f acc[2][2];
  #pragma unroll
  for (int i = 0; i < 2; ++i)
    #pragma unroll
    for (int j = 0; j < 2; ++j)
      #pragma unroll
      for (int r = 0; r < 8; ++r) acc[i][j][r] = 0.f;

  for (int k0 = 0; k0 < Kpad; k0 += 32) {
    __syncthreads();
    // ---- stage A: 128 x 32
    if constexpr (IN_F32) {
      // scalar path (conv1: fp32 input, CIN=51 not 32-aligned), 16 elems/thread
      #pragma unroll
      for (int i = 0; i < 16; ++i) {
        int e  = tid * 16 + i;
        int mr = e >> 5;
        int kk = e & 31;
        int k  = k0 + kk;
        int m  = m0 + mr;
        int bb = m / TLOC, t = m % TLOC;
        unsigned short hv = 0;
        if (k < K_REAL) {
          int rr = k / CIN, cc = k - rr * CIN;
          int gt = TMUL * t + ROW_OFF + rr;
          if (gt >= 0 && gt < TIN)
            hv = f2bf(inF[((size_t)bb * TIN + gt) * CIN + cc]);
        }
        As[mr * 40 + kk] = hv;
      }
    } else {
      // vector path: 32-wide K chunk lives in ONE window row -> 2 x b128
      const int mr  = tid >> 1;
      const int seg = (tid & 1) << 4;            // 0 or 16
      const int m   = m0 + mr;
      const int bb  = m / TLOC, t = m % TLOC;
      const int rr  = k0 / CIN;                  // chunk-constant
      const int cc0 = k0 - rr * CIN;
      const int gt  = TMUL * t + ROW_OFF + rr;
      unsigned short* dst = &As[mr * 40 + seg];
      if (gt >= 0 && gt < TIN) {
        const unsigned short* src = inH + ((size_t)bb * TIN + gt) * CIN + cc0 + seg;
        cp16_g2l(src, dst);
        cp16_g2l(src + 8, dst + 8);
      } else {
        uint4 z = make_uint4(0, 0, 0, 0);
        *(uint4*)dst = z;
        *(uint4*)(dst + 8) = z;
      }
    }
    // ---- stage B: 64 n-rows x 32 k, one aligned b128 per thread
    {
      const int nn = tid >> 2;
      const int ks = (tid & 3) << 3;             // 0,8,16,24
      cp16_g2l(Bp + (size_t)(n0 + nn) * Kpad + k0 + ks, &Bs[nn * 40 + ks]);
    }
    cp_wait();
    __syncthreads();

    // ---- fragments in contiguous k order (k = 16*half + s, A and B alike), then 4 WMMAs
    Frag16 af[2], bfr[2];
    #pragma unroll
    for (int i = 0; i < 2; ++i) {
      const unsigned short* ap = &As[(wm * 32 + i * 16 + l16) * 40];
      #pragma unroll
      for (int q = 0; q < 8; ++q) {
        int kv = 2 * q + 16 * half;                            // A 16x32 bf16
        af[i].u[q] = *(const uint32_t*)(ap + kv);
      }
    }
    #pragma unroll
    for (int j = 0; j < 2; ++j) {
      const unsigned short* bp = &Bs[(wn * 32 + j * 16 + l16) * 40];
      #pragma unroll
      for (int q = 0; q < 8; ++q) {
        int kv = 2 * q + 16 * half;                            // B 32x16 bf16
        bfr[j].u[q] = *(const uint32_t*)(bp + kv);
      }
    }
    #pragma unroll
    for (int i = 0; i < 2; ++i)
      #pragma unroll
      for (int j = 0; j < 2; ++j)
        acc[i][j] = wmma_bf16(af[i].v, bfr[j].v, acc[i][j]);
  }

  // ---- store C (fp32), parity-interleaved time for deconvs
  #pragma unroll
  for (int i = 0; i < 2; ++i)
    #pragma unroll
    for (int j = 0; j < 2; ++j)
      #pragma unroll
      for (int r = 0; r < 8; ++r) {
        int m = m0 + wm * 32 + i * 16 + r + 8 * half;
        int n = n0 + wn * 32 + j * 16 + l16;
        if (n < Nreal) {
          int bb = m / TLOC, t = m % TLOC;
          int tout = OUT_T_MUL * t + OUT_T_OFF;
          float v = acc[i][j][r];
          if (bias) v += bias[n];
          out[((size_t)bb * (TLOC * OUT_T_MUL) + tout) * Nreal + n] = v;
        }
      }
}

// ---------------------------------------------------------------------------
// BatchNorm batch statistics: per-channel sum & sumsq over M rows of (M, C).
// One block per channel.
// ---------------------------------------------------------------------------
__global__ __launch_bounds__(256) void bn_stats(
    const float* __restrict__ y, int Mtot, int C, float* __restrict__ stats)
{
  __shared__ float s1[256], s2[256];
  int c = blockIdx.x, tid = threadIdx.x;
  float a = 0.f, b = 0.f;
  for (int m = tid; m < Mtot; m += 256) {
    float v = y[(size_t)m * C + c];
    a += v; b += v * v;
  }
  s1[tid] = a; s2[tid] = b;
  __syncthreads();
  for (int s = 128; s > 0; s >>= 1) {
    if (tid < s) { s1[tid] += s1[tid + s]; s2[tid] += s2[tid + s]; }
    __syncthreads();
  }
  if (tid == 0) { stats[c] = s1[0]; stats[C + c] = s2[0]; }
}

// ---------------------------------------------------------------------------
// Fused BN(train-mode) + ReLU + bf16 cast; optional transposed fp32 output
// (writes z in (B, C, T') layout for the harness).
// ---------------------------------------------------------------------------
template<bool TRANS_OUT>
__global__ __launch_bounds__(256) void bn_relu_apply(
    const float* __restrict__ y, const float* __restrict__ stats,
    const float* __restrict__ gamma, const float* __restrict__ beta,
    unsigned short* __restrict__ out_bf, float* __restrict__ out_f32,
    int Mtot, int C, int Tper)
{
  size_t total = (size_t)Mtot * C;
  size_t step  = (size_t)gridDim.x * 256;
  float invM   = 1.f / (float)Mtot;
  for (size_t i = (size_t)blockIdx.x * 256 + threadIdx.x; i < total; i += step) {
    int c = (int)(i % C);
    float mean = stats[c] * invM;
    float var  = stats[C + c] * invM - mean * mean;   // biased variance
    float sc = gamma[c] * rsqrtf(var + 1e-5f);
    float sh = beta[c] - mean * sc;
    float v  = fmaxf(y[i] * sc + sh, 0.f);
    out_bf[i] = f2bf(v);
    if constexpr (TRANS_OUT) {
      size_t m = i / C;
      int bb = (int)(m / Tper), t = (int)(m % Tper);
      out_f32[((size_t)bb * C + c) * Tper + t] = v;
    }
  }
}

// ---------------------------------------------------------------------------
// VQ: codebook prep (bf16 [code][dim] pack + ||c||^2)
// ---------------------------------------------------------------------------
__global__ __launch_bounds__(256) void prep_codebook(
    const float* __restrict__ cb, unsigned short* __restrict__ cbB,
    float* __restrict__ cnorm)
{
  int n = blockIdx.x * 256 + threadIdx.x;
  if (n >= 512) return;
  float s = 0.f;
  for (int d = 0; d < 128; ++d) {
    float v = cb[(size_t)n * 128 + d];
    s += v * v;
    cbB[(size_t)n * 128 + d] = f2bf(v);   // [N=512 codes][K=128 dims]
  }
  cnorm[n] = s;
}

// ---------------------------------------------------------------------------
// VQ argmin: per wave 16 z-rows (K=128) x all 512 codes via WMMA;
// score = ||c||^2 - 2 z.c ; running argmin via 16x16 LDS spill per tile.
// Block = 8 waves = 128 rows.  Bs stride 136 hw: 16B-aligned rows (272B) and
// conflict-free fragment reads (68 dwords -> n*4 mod 64 distinct, 16 rows).
// ---------------------------------------------------------------------------
__global__ __launch_bounds__(256) void vq_argmin(
    const unsigned short* __restrict__ zbf, const unsigned short* __restrict__ cbB,
    const float* __restrict__ cnorm, int* __restrict__ idx_out)
{
  __shared__ unsigned short Bs[16 * 136];  // [code][dim]
  __shared__ float Ssc[8][16][17];

  const int tid  = threadIdx.x;
  const int lane = tid & 31, wave = tid >> 5;
  const int half = lane >> 4, l16 = lane & 15;
  const int rowbase = blockIdx.x * 128 + wave * 16;

  // A fragments: 16 rows x K=128 (4 k-steps), straight from global (B,T,D) bf16
  Frag16 a[4];
  {
    const unsigned short* zr = zbf + (size_t)(rowbase + l16) * 128;
    #pragma unroll
    for (int kk = 0; kk < 4; ++kk)
      #pragma unroll
      for (int q = 0; q < 8; ++q) {
        int kv = 2 * q + 16 * half + 32 * kk;
        a[kk].u[q] = *(const uint32_t*)(zr + kv);
      }
  }

  float bestv = 3.4e38f; int besti = 0;
  for (int nt = 0; nt < 32; ++nt) {          // 32 tiles of 16 codes
    __syncthreads();
    {                                        // stage 16 codes x 128 dims, b128
      int nl = tid >> 4;
      int ks = (tid & 15) << 3;
      cp16_g2l(cbB + (size_t)(nt * 16 + nl) * 128 + ks, &Bs[nl * 136 + ks]);
    }
    cp_wait();
    __syncthreads();

    v8f acc;
    #pragma unroll
    for (int r = 0; r < 8; ++r) acc[r] = 0.f;
    const unsigned short* bp = &Bs[l16 * 136];
    #pragma unroll
    for (int kk = 0; kk < 4; ++kk) {
      Frag16 bfr;
      #pragma unroll
      for (int q = 0; q < 8; ++q) {
        int kv = 2 * q + 16 * half + 32 * kk;
        bfr.u[q] = *(const uint32_t*)(bp + kv);
      }
      acc = wmma_bf16(a[kk].v, bfr.v, acc);
    }
    float cn = cnorm[nt * 16 + l16];         // column = l16
    #pragma unroll
    for (int r = 0; r < 8; ++r)
      Ssc[wave][r + 8 * half][l16] = cn - 2.f * acc[r];
    __syncthreads();
    if (half == 0) {                         // lane = row, scan 16 cols
      #pragma unroll
      for (int n = 0; n < 16; ++n) {
        float v = Ssc[wave][l16][n];
        if (v < bestv) { bestv = v; besti = nt * 16 + n; }  // first-min wins
      }
    }
  }
  if (half == 0) idx_out[rowbase + l16] = besti;
}

// ---------------------------------------------------------------------------
// Gather z_q = codebook[idx]: bf16 (B,T,D) for decoder, fp32 (B,D,T) output,
// indices-as-float output.
// ---------------------------------------------------------------------------
__global__ __launch_bounds__(128) void vq_gather(
    const int* __restrict__ idx, const float* __restrict__ cb,
    unsigned short* __restrict__ zq_bf, float* __restrict__ zq_out,
    float* __restrict__ idx_f)
{
  int row = blockIdx.x, d = threadIdx.x;   // row = b*1024 + t, d in [0,128)
  int id  = idx[row];
  float v = cb[(size_t)id * 128 + d];
  zq_bf[(size_t)row * 128 + d] = f2bf(v);
  int bb = row >> 10, t = row & 1023;
  zq_out[((size_t)bb * 128 + d) * 1024 + t] = v;
  if (d == 0) idx_f[row] = (float)id;
}

// ---------------------------------------------------------------------------
// Launcher
// ---------------------------------------------------------------------------
extern "C" void kernel_launch(void* const* d_in, const int* in_sizes, int n_in,
                              void* d_out, int out_size, void* d_ws, size_t ws_size,
                              hipStream_t stream)
{
  (void)in_sizes; (void)n_in; (void)out_size; (void)ws_size;

  const float* x    = (const float*)d_in[0];
  const float* w1   = (const float*)d_in[1];
  const float* g1   = (const float*)d_in[3];
  const float* be1  = (const float*)d_in[4];
  const float* w2   = (const float*)d_in[5];
  const float* g2   = (const float*)d_in[7];
  const float* be2  = (const float*)d_in[8];
  const float* cb   = (const float*)d_in[9];
  const float* dw1  = (const float*)d_in[10];
  const float* dg1  = (const float*)d_in[12];
  const float* dbe1 = (const float*)d_in[13];
  const float* dw2  = (const float*)d_in[14];
  const float* db2  = (const float*)d_in[15];
  // b1,b2,db1 unused: training-mode BN cancels preceding conv bias (all zero anyway)

  constexpr int T1 = 2048, T2 = 1024;
  constexpr int M1 = 128 * T1;   // 262144
  constexpr int M2 = 128 * T2;   // 131072

  // Output regions (all float per harness convention)
  float* xrec  = (float*)d_out;                 // (128,4096,51)
  float* zout  = xrec + (size_t)26738688;       // (128,128,1024)
  float* zqout = zout + (size_t)16777216;       // (128,128,1024)
  float* idxf  = zqout + (size_t)16777216;      // (128,1024)

  // Workspace bump allocator (reuses big fp32/bf16 buffers across phases)
  char* ws = (char*)d_ws;
  size_t off = 0;
  auto alloc = [&](size_t bytes) { void* p = ws + off; off += (bytes + 255) & ~size_t(255); return p; };
  float*          y1    = (float*)alloc((size_t)M1 * 256 * 4); // conv1 out; reused as deconv1 out
  unsigned short* h1    = (unsigned short*)alloc((size_t)M1 * 256 * 2); // act; reused as decoder act
  float*          y2    = (float*)alloc((size_t)M2 * 128 * 4);
  unsigned short* zbf   = (unsigned short*)alloc((size_t)M2 * 128 * 2);
  unsigned short* zqbf  = (unsigned short*)alloc((size_t)M2 * 128 * 2);
  unsigned short* cbB   = (unsigned short*)alloc(512 * 128 * 2);
  float*          cnorm = (float*)alloc(512 * 4);
  int*            idxb  = (int*)alloc((size_t)M2 * 4);
  float*          stats = (float*)alloc(2 * 256 * 4);
  unsigned short* Bw1   = (unsigned short*)alloc(256 * 224 * 2);
  unsigned short* Bw2   = (unsigned short*)alloc(128 * 1024 * 2);
  unsigned short* Bd1e  = (unsigned short*)alloc(256 * 256 * 2);
  unsigned short* Bd1o  = (unsigned short*)alloc(256 * 256 * 2);
  unsigned short* Bd2e  = (unsigned short*)alloc(64 * 512 * 2);
  unsigned short* Bd2o  = (unsigned short*)alloc(64 * 512 * 2);

  // ---- weight packing ([N][K]-major; taps byte r = kernel tap for window-row r)
  pack_weights<<<(256 * 224 + 255) / 256, 256, 0, stream>>>(w1,  Bw1, 256, 256,  51, 4, 4,  224, 0x03020100, 0);
  pack_weights<<<(128 * 1024 + 255) / 256, 256, 0, stream>>>(w2, Bw2, 128, 128, 256, 4, 4, 1024, 0x03020100, 0);
  pack_weights<<<(256 * 256 + 255) / 256, 256, 0, stream>>>(dw1, Bd1e, 256, 256, 128, 4, 2, 256, 0x0103, 1); // even: {W3,W1}
  pack_weights<<<(256 * 256 + 255) / 256, 256, 0, stream>>>(dw1, Bd1o, 256, 256, 128, 4, 2, 256, 0x0002, 1); // odd : {W2,W0}
  pack_weights<<<(64 * 512 + 255) / 256, 256, 0, stream>>>(dw2, Bd2e,  51,  64, 256, 4, 2, 512, 0x0103, 1);
  pack_weights<<<(64 * 512 + 255) / 256, 256, 0, stream>>>(dw2, Bd2o,  51,  64, 256, 4, 2, 512, 0x0002, 1);
  prep_codebook<<<2, 256, 0, stream>>>(cb, cbB, cnorm);

  // ---- encoder conv1: x (128,4096,51) f32 -> y1 (128,2048,256) f32
  conv_gemm<true, 51, 4, 2, -1, 1, 0, 4096, 2048>
      <<<dim3(M1 / 128, 4), 256, 0, stream>>>(x, Bw1, y1, nullptr, 256, 224);
  bn_stats<<<256, 256, 0, stream>>>(y1, M1, 256, stats);
  bn_relu_apply<false><<<8192, 256, 0, stream>>>(y1, stats, g1, be1, h1, nullptr, M1, 256, T1);

  // ---- encoder conv2: h1 bf16 -> y2 (128,1024,128) f32
  conv_gemm<false, 256, 4, 2, -1, 1, 0, 2048, 1024>
      <<<dim3(M2 / 128, 2), 256, 0, stream>>>(h1, Bw2, y2, nullptr, 128, 1024);
  bn_stats<<<128, 256, 0, stream>>>(y2, M2, 128, stats);
  // z (fp32, transposed to (B,D,T')) + zbf for VQ/decoder
  bn_relu_apply<true><<<8192, 256, 0, stream>>>(y2, stats, g2, be2, zbf, zout, M2, 128, T2);

  // ---- vector quantization
  vq_argmin<<<M2 / 128, 256, 0, stream>>>(zbf, cbB, cnorm, idxb);
  vq_gather<<<M2, 128, 0, stream>>>(idxb, cb, zqbf, zqout, idxf);

  // ---- decoder deconv1 (parity split): zqbf (128,1024,128) -> y1 (128,2048,256)
  conv_gemm<false, 128, 2, 1, -1, 2, 0, 1024, 1024>
      <<<dim3(M2 / 128, 4), 256, 0, stream>>>(zqbf, Bd1e, y1, nullptr, 256, 256);
  conv_gemm<false, 128, 2, 1, 0, 2, 1, 1024, 1024>
      <<<dim3(M2 / 128, 4), 256, 0, stream>>>(zqbf, Bd1o, y1, nullptr, 256, 256);
  bn_stats<<<256, 256, 0, stream>>>(y1, M1, 256, stats);
  bn_relu_apply<false><<<8192, 256, 0, stream>>>(y1, stats, dg1, dbe1, h1, nullptr, M1, 256, T1);

  // ---- decoder deconv2 (parity split) + bias -> x_recon (128,4096,51) directly
  conv_gemm<false, 256, 2, 1, -1, 2, 0, 2048, 2048>
      <<<dim3(M1 / 128, 1), 256, 0, stream>>>(h1, Bd2e, xrec, db2, 51, 512);
  conv_gemm<false, 256, 2, 1, 0, 2, 1, 2048, 2048>
      <<<dim3(M1 / 128, 1), 256, 0, stream>>>(h1, Bd2o, xrec, db2, 51, 512);
}
